// Graph_Generator_7395933684256
// MI455X (gfx1250) — matrix operand order repaired and checked by reading, not compiled, not run on hardware
//
#include <hip/hip_runtime.h>

typedef __attribute__((ext_vector_type(16))) _Float16 v16h;
typedef __attribute__((ext_vector_type(8)))  float    v8f;

#define NV   3      // views
#define NI   8192   // instances
#define ND   128    // feature dim
#define NL   64     // labels
#define KNN  16     // neighbors
#define WPB  4      // waves per block (main kernel)
#define NB   2      // 16-instance column blocks per wave

// ---------------- prep: f32 -> f16 conversion (coalesced, 8 elems/thread) ----------------
__global__ void cvt_f16_kernel(const float* __restrict__ in, _Float16* __restrict__ out, int n8) {
    int t = blockIdx.x * blockDim.x + threadIdx.x;
    if (t >= n8) return;
    const float4* p = (const float4*)in + (size_t)t * 2;
    float4 a = p[0], b = p[1];
    _Float16 h[8] = {(_Float16)a.x, (_Float16)a.y, (_Float16)a.z, (_Float16)a.w,
                     (_Float16)b.x, (_Float16)b.y, (_Float16)b.z, (_Float16)b.w};
    *(float4*)(out + (size_t)t * 8) = *(const float4*)h;
}

// ---------------- prep: squared norms, one wave32 per row ----------------
__global__ void sqnorm_kernel(const float* __restrict__ feats, float* __restrict__ sq, int rows) {
    int w    = (blockIdx.x * blockDim.x + threadIdx.x) >> 5;
    int lane = threadIdx.x & 31;
    if (w >= rows) return;
    float4 x = ((const float4*)(feats + (size_t)w * ND))[lane];  // 32 lanes x 4 floats = 128
    float s = x.x * x.x + x.y * x.y + x.z * x.z + x.w * x.w;
    #pragma unroll
    for (int off = 16; off; off >>= 1) s += __shfl_xor(s, off, 32);
    if (lane == 0) sq[w] = s;
}

// ---------------- main: fused WMMA Gram + streaming top-16 ----------------
// One wave owns NB*16 instances (WMMA B / columns, register-resident); streams all 8192
// candidates through WMMA A. Accumulator layout (16x16 f32): lane l holds column
// n = nbase + b*16 + (l&15); rows m = mbase + (l>>4)*8 + r for r = 0..7 in c[b][r].
// Per-lane sorted top-16 lists live in LDS (dynamic indexing, non-unrolled shift loop);
// a register threshold kd15 gates the rare insertion path.
__global__ __launch_bounds__(WPB * 32)
void knn_wmma_kernel(const _Float16* __restrict__ f16, const float* __restrict__ sq,
                     float* __restrict__ topk_dist, long long* __restrict__ edges) {
    __shared__ float s_d[WPB][NB][32][KNN];   // 16 KB
    __shared__ int   s_i[WPB][NB][32][KNN];   // 16 KB

    const int lane = threadIdx.x & 31;
    const int wid  = threadIdx.x >> 5;
    const int gw   = blockIdx.x * WPB + wid;            // global wave id
    const int bpv  = NI / (16 * NB);                    // wave-blocks per view = 256
    const int v     = gw / bpv;
    const int nbase = (gw % bpv) * (16 * NB);
    const int l15   = lane & 15;
    const int hb    = lane >> 4;

    const _Float16* fv = f16 + (size_t)v * NI * ND;
    const float*    sv = sq  + (size_t)v * NI;

    v16h  B[NB][4];
    float sq_n[NB], kd15[NB];
    int   n_my[NB];
    float* ld[NB];
    int*   li[NB];
    #pragma unroll
    for (int b = 0; b < NB; ++b) {
        const _Float16* pb = fv + (size_t)(nbase + b * 16 + l15) * ND + hb * 16;
        #pragma unroll
        for (int kb = 0; kb < 4; ++kb) {
            *(float4*)&B[b][kb]       = *(const float4*)(pb + kb * 32);
            *((float4*)&B[b][kb] + 1) = *(const float4*)(pb + kb * 32 + 8);
        }
        sq_n[b] = sv[nbase + b * 16 + l15];
        n_my[b] = nbase + b * 16 + l15;
        kd15[b] = 3.4e38f;
        ld[b] = &s_d[wid][b][lane][0];
        li[b] = &s_i[wid][b][lane][0];
        #pragma clang loop unroll(disable)
        for (int i = 0; i < KNN; ++i) { ld[b][i] = 3.4e38f; li[b][i] = 0; }
    }

    for (int mbase = 0; mbase < NI; mbase += 16) {
        const _Float16* pa = fv + (size_t)(mbase + l15) * ND + hb * 16;
        __builtin_prefetch(pa + (size_t)16 * ND, 0, 3);   // WGP-scope prefetch of next tile

        v16h A[4];
        #pragma unroll
        for (int kb = 0; kb < 4; ++kb) {
            *(float4*)&A[kb]       = *(const float4*)(pa + kb * 32);
            *((float4*)&A[kb] + 1) = *(const float4*)(pa + kb * 32 + 8);
        }

        v8f c[NB];
        #pragma unroll
        for (int b = 0; b < NB; ++b) c[b] = (v8f){};
        #pragma unroll
        for (int kb = 0; kb < 4; ++kb)
            #pragma unroll
            for (int b = 0; b < NB; ++b)   // independent accumulators back-to-back
                c[b] = __builtin_amdgcn_wmma_f32_16x16x32_f16(false, A[kb], false, B[b][kb],
                                                              (short)0, c[b], false, false);

        float smv = sv[mbase + l15];       // candidate norms, broadcast via lane permute
        #pragma unroll
        for (int r = 0; r < 8; ++r) {
            const int   m  = mbase + hb * 8 + r;
            const float sm = __shfl(smv, hb * 8 + r, 32);
            #pragma unroll
            for (int b = 0; b < NB; ++b) {
                float d2 = sq_n[b] + sm - 2.0f * c[b][r];
                if (m == n_my[b]) d2 = 0.0f;               // exact zero on diagonal
                if (d2 < kd15[b]) {                        // rare: insert into LDS list
                    int pos = KNN - 1;
                    #pragma clang loop unroll(disable)
                    for (; pos > 0; --pos) {               // compact dynamic shift loop
                        float pv = ld[b][pos - 1];
                        if (!(pv > d2)) break;
                        ld[b][pos] = pv;
                        li[b][pos] = li[b][pos - 1];
                    }
                    ld[b][pos] = d2;
                    li[b][pos] = m;
                    kd15[b] = ld[b][KNN - 1];
                }
            }
        }
    }

    __syncthreads();   // cross-lane LDS visibility for the merge

    // merge half-lists (lane n <-> lane n+16) per column block; write outputs
    if (lane < 16) {
        #pragma unroll 1
        for (int b = 0; b < NB; ++b) {
            const int n = nbase + b * 16 + lane;
            const float* da  = &s_d[wid][b][lane][0];
            const int*   iav = &s_i[wid][b][lane][0];
            const float* db  = &s_d[wid][b][lane + 16][0];
            const int*   ibv = &s_i[wid][b][lane + 16][0];
            const size_t rowo = ((size_t)v * NI + n) * KNN;
            long long* srcp = edges + (size_t)v * 2 * NI * KNN;   // [V,2,N*k] plane 0
            long long* dstp = srcp + (size_t)NI * KNN;            // plane 1
            int ia = 0, ib = 0;
            #pragma clang loop unroll(disable)
            for (int t = 0; t < KNN; ++t) {    // ia+ib==t<=15 -> always in bounds
                float x0 = da[ia], x1 = db[ib];
                bool  ta = x0 <= x1;
                float d  = ta ? x0 : x1;
                int   id = ta ? iav[ia] : ibv[ib];
                ia += ta ? 1 : 0;
                ib += ta ? 0 : 1;
                topk_dist[rowo + t]       = sqrtf(fmaxf(d, 0.0f));
                srcp[(size_t)n * KNN + t] = (long long)n;
                dstp[(size_t)n * KNN + t] = (long long)id;
            }
        }
    }
}

// ---------------- cross-edge index patterns ----------------
__global__ void cross_view_kernel(long long* __restrict__ out, int total) {  // total = N*V
    int t = blockIdx.x * blockDim.x + threadIdx.x;
    if (t >= total) return;
    out[2 * t]     = (long long)(t / NV);
    out[2 * t + 1] = (long long)(t % NV);
}
__global__ void cross_label_kernel(long long* __restrict__ out, int total) { // total = N*L
    int t = blockIdx.x * blockDim.x + threadIdx.x;
    if (t >= total) return;
    out[2 * t]     = (long long)(t / NL);
    out[2 * t + 1] = (long long)(t % NL);
}

extern "C" void kernel_launch(void* const* d_in, const int* in_sizes, int n_in,
                              void* d_out, int out_size, void* d_ws, size_t ws_size,
                              hipStream_t stream) {
    (void)in_sizes; (void)n_in; (void)out_size; (void)ws_size;
    const float* train = (const float*)d_in[0];   // [V,N,D] f32
    const float* label = (const float*)d_in[1];   // [L,D]   f32
    // d_in[2]=target (unused), d_in[3]=k (fixed 16)

    // workspace: f16 features + squared norms
    _Float16* f16 = (_Float16*)d_ws;
    float*    sq  = (float*)((char*)d_ws + (size_t)NV * NI * ND * sizeof(_Float16));

    // output regions (concatenated, return order)
    char* out = (char*)d_out;
    size_t off = 0;
    float*     o_topk  = (float*)(out + off);      off += (size_t)NV * NI * KNN * 4;
    long long* o_edges = (long long*)(out + off);  off += (size_t)NV * 2 * NI * KNN * 8;
    float*     o_train = (float*)(out + off);      off += (size_t)NV * NI * ND * 4;
    float*     o_label = (float*)(out + off);      off += (size_t)NL * ND * 4;
    long long* o_view  = (long long*)(out + off);  off += (size_t)NI * NV * 2 * 8;
    long long* o_glab  = (long long*)(out + off);

    // prep
    {
        int n8 = NV * NI * ND / 8;
        cvt_f16_kernel<<<(n8 + 255) / 256, 256, 0, stream>>>(train, f16, n8);
        int rows = NV * NI;
        sqnorm_kernel<<<(rows * 32 + 255) / 256, 256, 0, stream>>>(train, sq, rows);
    }

    // fused WMMA Gram + top-16: 768 independent waves, 4 waves/block -> 192 WGs
    knn_wmma_kernel<<<(NV * (NI / (16 * NB))) / WPB, WPB * 32, 0, stream>>>(f16, sq, o_topk, o_edges);

    // cross-edge patterns
    {
        int tv = NI * NV;
        cross_view_kernel<<<(tv + 255) / 256, 256, 0, stream>>>(o_view, tv);
        int tl = NI * NL;
        cross_label_kernel<<<(tl + 255) / 256, 256, 0, stream>>>(o_glab, tl);
    }

    // passthrough copies (graph-capture safe d2d async copies)
    hipMemcpyAsync(o_train, train, (size_t)NV * NI * ND * 4, hipMemcpyDeviceToDevice, stream);
    hipMemcpyAsync(o_label, label, (size_t)NL * ND * 4, hipMemcpyDeviceToDevice, stream);
}
